// PointNet_10866267258988
// MI455X (gfx1250) — matrix operand order repaired and checked by reading, not compiled, not run on hardware
//
#include <hip/hip_runtime.h>
#include <hip/hip_bf16.h>
#include <cstdint>
#include <cstddef>

typedef __attribute__((ext_vector_type(16))) _Float16 v16h;
typedef __attribute__((ext_vector_type(8)))  _Float16 v8h;
typedef __attribute__((ext_vector_type(8)))  float    v8f;
typedef __attribute__((ext_vector_type(4)))  int      v4i;

#define PN_GAS __attribute__((address_space(1)))
#define PN_LAS __attribute__((address_space(3)))

#if defined(__HIP_DEVICE_COMPILE__) && __has_builtin(__builtin_amdgcn_global_load_async_to_lds_b128)
#define PN_HAVE_ASYNC 1
#else
#define PN_HAVE_ASYNC 0
#endif

// ---------------------------------------------------------------------------
// Zero-fill workspace (graph-capture-safe alternative to hipMemsetAsync)
// ---------------------------------------------------------------------------
__global__ void pn_zero_kernel(uint32_t* p, size_t n) {
    size_t i = (size_t)blockIdx.x * blockDim.x + threadIdx.x;
    if (i < n) p[i] = 0u;
}

// ---------------------------------------------------------------------------
// Fold BN into scale/bias, convert weights to f16 with K padded to 32
// ---------------------------------------------------------------------------
__global__ void pn_prep_layer_kernel(const float* __restrict__ w,
                                     const float* __restrict__ b,
                                     const float* __restrict__ g,
                                     const float* __restrict__ be,
                                     const float* __restrict__ mu,
                                     const float* __restrict__ var,
                                     _Float16* __restrict__ wh,
                                     float* __restrict__ S, float* __restrict__ T,
                                     int cout, int cin, int kpad) {
    int i = blockIdx.x * blockDim.x + threadIdx.x;
    if (i < cout) {
        float s = g[i] * rsqrtf(var[i] + 1e-5f);
        S[i] = s;
        T[i] = be[i] + (b[i] - mu[i]) * s;
    }
    int total = cout * kpad;
    if (i < total) {
        int c = i / kpad, k = i % kpad;
        wh[i] = (_Float16)((k < cin) ? w[c * cin + k] : 0.0f);
    }
}

__global__ void pn_convert_h_kernel(const float* __restrict__ src,
                                    _Float16* __restrict__ dst, int n) {
    int i = blockIdx.x * blockDim.x + threadIdx.x;
    if (i < n) dst[i] = (_Float16)src[i];
}

// ---------------------------------------------------------------------------
// Farthest point sampling: one block per batch, iterative argmax reduction
// (matches jnp.argmax first-max tie-break: keep lowest index on ties)
// ---------------------------------------------------------------------------
#define FPS_T 512
__global__ void pn_fps_kernel(const float* __restrict__ xyz, // [B][N][3]
                              float* __restrict__ dist,      // [B][N]
                              int* __restrict__ outIdx,      // [B][npoint]
                              int N, int npoint) {
    int b = blockIdx.x;
    const float* p = xyz + (size_t)b * N * 3;
    float* d = dist + (size_t)b * N;
    __shared__ float sval[FPS_T];
    __shared__ int   sidx[FPS_T];
    __shared__ int   far_s;
    for (int i = threadIdx.x; i < N; i += FPS_T) d[i] = 1e10f;
    if (threadIdx.x == 0) far_s = 0;
    __syncthreads();
    int far = 0;
    for (int it = 0; it < npoint; ++it) {
        if (threadIdx.x == 0) outIdx[(size_t)b * npoint + it] = far;
        float cx = p[far * 3 + 0], cy = p[far * 3 + 1], cz = p[far * 3 + 2];
        float bestv = -1.0f; int besti = N - 1;
        for (int i = threadIdx.x; i < N; i += FPS_T) {
            float dx = p[i * 3 + 0] - cx;
            float dy = p[i * 3 + 1] - cy;
            float dz = p[i * 3 + 2] - cz;
            float dd = dx * dx + dy * dy + dz * dz;
            float nd = fminf(d[i], dd);
            d[i] = nd;
            if (nd > bestv) { bestv = nd; besti = i; }  // ascending i keeps first max
        }
        sval[threadIdx.x] = bestv; sidx[threadIdx.x] = besti;
        __syncthreads();
        for (int s = FPS_T / 2; s > 0; s >>= 1) {
            if (threadIdx.x < s) {
                float v2 = sval[threadIdx.x + s]; int i2 = sidx[threadIdx.x + s];
                if (v2 > sval[threadIdx.x] ||
                    (v2 == sval[threadIdx.x] && i2 < sidx[threadIdx.x])) {
                    sval[threadIdx.x] = v2; sidx[threadIdx.x] = i2;
                }
            }
            __syncthreads();
        }
        if (threadIdx.x == 0) far_s = sidx[0];
        __syncthreads();
        far = far_s;
    }
}

__global__ void pn_gather_xyz_kernel(const float* __restrict__ xyz,
                                     const int* __restrict__ idx,
                                     float* __restrict__ out,
                                     int B, int N, int S) {
    int t = blockIdx.x * blockDim.x + threadIdx.x;
    if (t >= B * S) return;
    int b = t / S;
    int id = idx[t]; if (id >= N) id = N - 1; if (id < 0) id = 0;
    const float* p = xyz + ((size_t)b * N + id) * 3;
    out[t * 3 + 0] = p[0]; out[t * 3 + 1] = p[1]; out[t * 3 + 2] = p[2];
}

// ---------------------------------------------------------------------------
// Ball query: first nsample(=16) points within radius, pad with first found
// ---------------------------------------------------------------------------
__global__ void pn_ball_kernel(const float* __restrict__ xyz,   // [B][N][3]
                               const float* __restrict__ nxyz,  // [B][S][3]
                               int* __restrict__ gi,            // [B][S][16]
                               int B, int N, int S, float r2) {
    int t = blockIdx.x * blockDim.x + threadIdx.x;
    if (t >= B * S) return;
    int b = t / S;
    const float* c = nxyz + (size_t)t * 3;
    float cx = c[0], cy = c[1], cz = c[2];
    const float* p = xyz + (size_t)b * N * 3;
    int* o = gi + (size_t)t * 16;
    int cnt = 0; int first = N - 1;
    for (int i = 0; i < N && cnt < 16; ++i) {
        float dx = p[i * 3 + 0] - cx;
        float dy = p[i * 3 + 1] - cy;
        float dz = p[i * 3 + 2] - cz;
        if (dx * dx + dy * dy + dz * dz <= r2) {
            if (cnt == 0) first = i;
            o[cnt++] = i;
        }
    }
    for (int k = cnt; k < 16; ++k) o[k] = first;
}

// ---------------------------------------------------------------------------
// Grouping: build X[(b*S+s)*16+j][kpad] = [xyz-center (3) | feat (C) | 0-pad]
// ---------------------------------------------------------------------------
__global__ void pn_group_kernel(const float* __restrict__ xyz,      // [B][N][3]
                                const float* __restrict__ nxyz,     // [B][S][3]
                                const int* __restrict__ gi,         // [B][S][16]
                                const _Float16* __restrict__ feat,  // [B*N][C]
                                _Float16* __restrict__ X,           // [B*S*16][kpad]
                                int B, int N, int S, int C, int kpad) {
    int t = blockIdx.x * blockDim.x + threadIdx.x;
    int total = B * S * 16;
    if (t >= total) return;
    int bs = t >> 4;
    int b = bs / S, s = bs % S;
    int id = gi[t]; if (id >= N) id = N - 1; if (id < 0) id = 0;
    const float* pp = xyz + ((size_t)b * N + id) * 3;
    const float* cc = nxyz + ((size_t)b * S + s) * 3;
    _Float16* row = X + (size_t)t * kpad;
    row[0] = (_Float16)(pp[0] - cc[0]);
    row[1] = (_Float16)(pp[1] - cc[1]);
    row[2] = (_Float16)(pp[2] - cc[2]);
    const _Float16* f = feat + ((size_t)b * N + id) * C;
    for (int c = 0; c < C; ++c) row[3 + c] = f[c];
    for (int c = 3 + C; c < kpad; ++c) row[c] = (_Float16)0.0f;  // keep pads clean
}

// ---------------------------------------------------------------------------
// WMMA GEMM + folded BN (+ReLU): D[m][n] = relu(S[m]*(W·X)[m][n] + T[m])
//
// Block = 256 threads = 8 waves. blockIdx.y selects a 16-row cout tile whose
// weights (16 x Kpad <= 12 KB) are staged into LDS once per block — via
// GLOBAL_LOAD_ASYNC_TO_LDS_B128 (ASYNCcnt / s_wait_asynccnt) when available.
// Each wave then computes a 16(cout) x 64(points) slab: the A fragment is
// read from LDS (ds_load_b128) once per k-step and reused by 4 WMMAs, so the
// wmma : global-load ratio is 4 : 8 instead of 1 : 4.
//
// Fragment layouts used here (wave32; A/B use contiguous k, not the ISA's table order):
//   A lane L: m=L&15, half e     -> k=(L>>4)*16+e
//   B lane L: n=L&15, half e     -> k=(L>>4)*16+e   (contiguous 32B load)
//   D lane L: n=L&15, vgpr r     -> m=r+8*(L>>4)
// ---------------------------------------------------------------------------
__launch_bounds__(256)
__global__ void pn_gemm_bn_kernel(const _Float16* __restrict__ X,
                                  const _Float16* __restrict__ W,
                                  const float* __restrict__ S,
                                  const float* __restrict__ T,
                                  _Float16* __restrict__ Yh,  // [M][Cout] or null
                                  float* __restrict__ Yf,     // [M][Cout] or null
                                  int M, int Kpad, int Cout, int relu) {
    __shared__ _Float16 sW[16 * 384];  // max Kpad = 384
    const int lane = threadIdx.x & 31;
    const int wave = threadIdx.x >> 5;
    const int c0 = blockIdx.y * 16;

    // ---- stage this block's weight tile (rows c0..c0+15, contiguous) in LDS
    {
        const _Float16* wsrc = W + (size_t)c0 * Kpad;
        int chunks = (16 * Kpad) >> 3;           // 16-byte chunks
        for (int i = threadIdx.x; i < chunks; i += 256) {
#if PN_HAVE_ASYNC
            __builtin_amdgcn_global_load_async_to_lds_b128(
                (PN_GAS v4i*)(wsrc + (size_t)i * 8),
                (PN_LAS v4i*)(&sW[i * 8]), 0, 0);
#else
            *(v8h*)(&sW[i * 8]) = *(const v8h*)(wsrc + (size_t)i * 8);
#endif
        }
#if PN_HAVE_ASYNC
#if __has_builtin(__builtin_amdgcn_s_wait_asynccnt)
        __builtin_amdgcn_s_wait_asynccnt(0);
#else
        asm volatile("s_wait_asynccnt 0x0" ::: "memory");
#endif
#endif
        __syncthreads();
    }

    const int n0 = (blockIdx.x * 8 + wave) * 64;   // 64-point slab per wave
    if (n0 >= M) return;
    int nt = (M - n0) >> 4; if (nt > 4) nt = 4;    // valid 16-point sub-tiles
    const int nl = lane & 15;
    const int hi = lane >> 4;

    const _Float16* sWrow = sW + nl * Kpad;                          // A row m=nl
    const _Float16* xbase = X + (size_t)(n0 + nl) * Kpad + hi * 16;  // B base

    v8f acc0 = {}, acc1 = {}, acc2 = {}, acc3 = {};
    for (int k0 = 0; k0 < Kpad; k0 += 32) {
        if (k0 + 32 < Kpad)
            __builtin_prefetch((const void*)(xbase + k0 + 32), 0, 1);  // global_prefetch_b8
        v8h a0 = *(const v8h*)(sWrow + k0 + hi * 16);       // ds_load_b128
        v8h a1 = *(const v8h*)(sWrow + k0 + hi * 16 + 8);   // ds_load_b128
        v16h a;
        #pragma unroll
        for (int e = 0; e < 8; ++e) { a[e] = a0[e]; a[8 + e] = a1[e]; }
        v16h b0 = *(const v16h*)(xbase + k0);
        acc0 = __builtin_amdgcn_wmma_f32_16x16x32_f16(false, a, false, b0,
                                                      (short)0, acc0, false, false);
        if (nt > 1) {
            v16h b1 = *(const v16h*)(xbase + (size_t)16 * Kpad + k0);
            acc1 = __builtin_amdgcn_wmma_f32_16x16x32_f16(false, a, false, b1,
                                                          (short)0, acc1, false, false);
        }
        if (nt > 2) {
            v16h b2 = *(const v16h*)(xbase + (size_t)32 * Kpad + k0);
            acc2 = __builtin_amdgcn_wmma_f32_16x16x32_f16(false, a, false, b2,
                                                          (short)0, acc2, false, false);
        }
        if (nt > 3) {
            v16h b3 = *(const v16h*)(xbase + (size_t)48 * Kpad + k0);
            acc3 = __builtin_amdgcn_wmma_f32_16x16x32_f16(false, a, false, b3,
                                                          (short)0, acc3, false, false);
        }
    }

    auto emit = [&](const v8f& acc, int u) {
        int nn = n0 + u * 16 + nl;
        #pragma unroll
        for (int r = 0; r < 8; ++r) {
            int m = c0 + 8 * hi + r;
            float v = acc[r] * S[m] + T[m];
            if (relu) v = fmaxf(v, 0.0f);
            if (Yh) Yh[(size_t)nn * Cout + m] = (_Float16)v;
            if (Yf) Yf[(size_t)nn * Cout + m] = v;
        }
    };
    emit(acc0, 0);
    if (nt > 1) emit(acc1, 1);
    if (nt > 2) emit(acc2, 2);
    if (nt > 3) emit(acc3, 3);
}

// ---------------------------------------------------------------------------
// Max-pool over the 16 neighbors of each center
// ---------------------------------------------------------------------------
__global__ void pn_maxpool_kernel(const _Float16* __restrict__ Y,  // [rows*16][C]
                                  _Float16* __restrict__ out,      // [rows][C]
                                  int rows, int C) {
    int t = blockIdx.x * blockDim.x + threadIdx.x;
    if (t >= rows * C) return;
    int c = t % C, bs = t / C;
    float m = -1e30f;
    for (int j = 0; j < 16; ++j)
        m = fmaxf(m, (float)Y[((size_t)bs * 16 + j) * C + c]);
    out[(size_t)bs * C + c] = (_Float16)m;
}

// ---------------------------------------------------------------------------
// Skip-connection copy into concat buffer
// ---------------------------------------------------------------------------
__global__ void pn_copy_skip_kernel(const _Float16* __restrict__ src, // [rows][C]
                                    _Float16* __restrict__ X,         // [rows][stride]
                                    int rows, int C, int stride, int off) {
    int t = blockIdx.x * blockDim.x + threadIdx.x;
    if (t >= rows * C) return;
    int c = t % C, r = t / C;
    X[(size_t)r * stride + off + c] = src[t];
}

// ---------------------------------------------------------------------------
// 3-NN inverse-distance interpolation (feature propagation)
// ---------------------------------------------------------------------------
__global__ void pn_interp_kernel(const float* __restrict__ xyz1,   // [B][Na][3] target
                                 const float* __restrict__ xyz2,   // [B][Nb][3] source
                                 const _Float16* __restrict__ f2,  // [B*Nb][C]
                                 _Float16* __restrict__ X,         // [B*Na][stride]
                                 int B, int Na, int Nb, int C, int stride, int off) {
    int t = blockIdx.x * blockDim.x + threadIdx.x;
    if (t >= B * Na) return;
    int b = t / Na;
    const float* p = xyz1 + (size_t)t * 3;
    float px = p[0], py = p[1], pz = p[2];
    const float* q = xyz2 + (size_t)b * Nb * 3;
    float d0 = 1e30f, d1 = 1e30f, d2 = 1e30f;
    int i0 = 0, i1 = 0, i2 = 0;
    for (int i = 0; i < Nb; ++i) {
        float dx = q[i * 3 + 0] - px;
        float dy = q[i * 3 + 1] - py;
        float dz = q[i * 3 + 2] - pz;
        float d = dx * dx + dy * dy + dz * dz;
        if (d < d0)      { d2 = d1; i2 = i1; d1 = d0; i1 = i0; d0 = d; i0 = i; }
        else if (d < d1) { d2 = d1; i2 = i1; d1 = d; i1 = i; }
        else if (d < d2) { d2 = d; i2 = i; }
    }
    float w0 = 1.0f / (d0 + 1e-8f);
    float w1 = 1.0f / (d1 + 1e-8f);
    float w2 = 1.0f / (d2 + 1e-8f);
    float wsum = w0 + w1 + w2;
    w0 /= wsum; w1 /= wsum; w2 /= wsum;
    const _Float16* fa = f2 + ((size_t)b * Nb + i0) * C;
    const _Float16* fb = f2 + ((size_t)b * Nb + i1) * C;
    const _Float16* fc = f2 + ((size_t)b * Nb + i2) * C;
    _Float16* row = X + (size_t)t * stride + off;
    for (int c = 0; c < C; ++c)
        row[c] = (_Float16)(w0 * (float)fa[c] + w1 * (float)fb[c] + w2 * (float)fc[c]);
}

// ---------------------------------------------------------------------------
// Global mean over points per (batch, channel)
// ---------------------------------------------------------------------------
__global__ void pn_mean_kernel(const _Float16* __restrict__ f, // [B*N][C]
                               float* __restrict__ gc,         // [B][C]
                               int N, int C) {
    int c = blockIdx.x, b = blockIdx.y;
    __shared__ float ssum[256];
    float s = 0.0f;
    for (int n = threadIdx.x; n < N; n += 256)
        s += (float)f[((size_t)b * N + n) * C + c];
    ssum[threadIdx.x] = s;
    __syncthreads();
    for (int st = 128; st > 0; st >>= 1) {
        if (threadIdx.x < st) ssum[threadIdx.x] += ssum[threadIdx.x + st];
        __syncthreads();
    }
    if (threadIdx.x == 0) gc[b * C + c] = ssum[0] / (float)N;
}

// Build head input: [l0_p (C) | broadcast global mean (C)]
__global__ void pn_headx_kernel(const _Float16* __restrict__ l0p, // [B*N][C]
                                const float* __restrict__ gc,     // [B][C]
                                _Float16* __restrict__ X,         // [B*N][2C]
                                int B, int N, int C) {
    int t = blockIdx.x * blockDim.x + threadIdx.x;
    if (t >= B * N * C) return;
    int c = t % C, bn = t / C, b = bn / N;
    X[(size_t)bn * (2 * C) + c]     = l0p[(size_t)bn * C + c];
    X[(size_t)bn * (2 * C) + C + c] = (_Float16)gc[b * C + c];
}

// Final 1-channel head + output assembly [x,y,z,logit]
__global__ void pn_head2_kernel(const float* __restrict__ y,  // [B*N][K]
                                const float* __restrict__ w,  // [K]
                                const float* __restrict__ bb, // [1]
                                const float* __restrict__ points, // [B*N][3]
                                float* __restrict__ out,      // [B*N][4]
                                int BN, int K) {
    int t = blockIdx.x * blockDim.x + threadIdx.x;
    if (t >= BN) return;
    const float* r = y + (size_t)t * K;
    float s = bb[0];
    for (int k = 0; k < K; ++k) s += w[k] * r[k];
    out[t * 4 + 0] = points[t * 3 + 0];
    out[t * 4 + 1] = points[t * 3 + 1];
    out[t * 4 + 2] = points[t * 3 + 2];
    out[t * 4 + 3] = s;
}

// ---------------------------------------------------------------------------
// Host orchestration
// ---------------------------------------------------------------------------
extern "C" void kernel_launch(void* const* d_in, const int* in_sizes, int n_in,
                              void* d_out, int out_size, void* d_ws, size_t ws_size,
                              hipStream_t stream) {
    (void)in_sizes; (void)n_in; (void)out_size; (void)ws_size;
    const int B = 2, N0 = 15104, N1 = 3776, N2 = 944, N3 = 236;
    const int R0 = B * N0;

    // Layer table: 17 conv_bn layers, input index base = 2 + 6*l (w,b,gamma,beta,mean,var)
    static const int LCOUT[17] = {32,32,64, 64,64,128, 128,128,256, 256,256, 256,128, 128,128,128, 128};
    static const int LCIN [17] = { 4,32,32, 67,64, 64, 131,128,128, 384,256, 320,256, 128,128,128, 256};
    int KP[17];
    _Float16* wh[17]; float* Sv[17]; float* Tv[17];

    char* base = (char*)d_ws;
    size_t cur = 0;
    auto alloc = [&](size_t bytes) -> void* {
        cur = (cur + 255) & ~(size_t)255;
        void* p = base + cur;
        cur += bytes;
        return p;
    };
    for (int l = 0; l < 17; ++l) {
        KP[l] = ((LCIN[l] + 31) / 32) * 32;
        wh[l] = (_Float16*)alloc((size_t)LCOUT[l] * KP[l] * sizeof(_Float16));
        Sv[l] = (float*)alloc((size_t)LCOUT[l] * sizeof(float));
        Tv[l] = (float*)alloc((size_t)LCOUT[l] * sizeof(float));
    }
    _Float16* feat0 = (_Float16*)alloc((size_t)R0 * sizeof(_Float16));
    float* dist = (float*)alloc((size_t)R0 * sizeof(float));
    int*   idx1 = (int*)alloc((size_t)B * N1 * sizeof(int));
    float* xyz1 = (float*)alloc((size_t)B * N1 * 3 * sizeof(float));
    int*   bal1 = (int*)alloc((size_t)B * N1 * 16 * sizeof(int));
    int*   idx2 = (int*)alloc((size_t)B * N2 * sizeof(int));
    float* xyz2 = (float*)alloc((size_t)B * N2 * 3 * sizeof(float));
    int*   bal2 = (int*)alloc((size_t)B * N2 * 16 * sizeof(int));
    int*   idx3 = (int*)alloc((size_t)B * N3 * sizeof(int));
    float* xyz3 = (float*)alloc((size_t)B * N3 * 3 * sizeof(float));
    int*   bal3 = (int*)alloc((size_t)B * N3 * 16 * sizeof(int));
    const size_t XCAP = (size_t)8 * 1024 * 1024 + 64 * 384;  // halves (+ slab slack)
    _Float16* XA = (_Float16*)alloc(XCAP * sizeof(_Float16));
    _Float16* XB = (_Float16*)alloc(XCAP * sizeof(_Float16));
    _Float16* l1p = (_Float16*)alloc((size_t)B * N1 * 64  * sizeof(_Float16));
    _Float16* l2p = (_Float16*)alloc((size_t)B * N2 * 128 * sizeof(_Float16));
    _Float16* l3p = (_Float16*)alloc((size_t)B * N3 * 256 * sizeof(_Float16));
    _Float16* f3o = (_Float16*)alloc(((size_t)B * N2 + 64) * 256 * sizeof(_Float16));
    _Float16* f2o = (_Float16*)alloc((size_t)B * N1 * 128 * sizeof(_Float16));
    _Float16* l0p = (_Float16*)alloc((size_t)R0 * 128 * sizeof(_Float16));
    float* h1o = (float*)alloc((size_t)R0 * 128 * sizeof(float));
    float* gc  = (float*)alloc((size_t)B * 128 * sizeof(float));

    // Zero entire used workspace (padded K columns must read as 0)
    {
        size_t words = (cur + 3) / 4;
        pn_zero_kernel<<<(unsigned)((words + 255) / 256), 256, 0, stream>>>((uint32_t*)d_ws, words);
    }

    const float* points = (const float*)d_in[0];
    const float* feats  = (const float*)d_in[1];
    auto P = [&](int i) { return (const float*)d_in[i]; };

    for (int l = 0; l < 17; ++l) {
        int bi = 2 + 6 * l;
        int tot = LCOUT[l] * KP[l];
        pn_prep_layer_kernel<<<(tot + 255) / 256, 256, 0, stream>>>(
            P(bi), P(bi + 1), P(bi + 2), P(bi + 3), P(bi + 4), P(bi + 5),
            wh[l], Sv[l], Tv[l], LCOUT[l], LCIN[l], KP[l]);
    }
    pn_convert_h_kernel<<<(R0 + 255) / 256, 256, 0, stream>>>(feats, feat0, R0);

    auto gemm = [&](const _Float16* X, int l, _Float16* Yh, float* Yf, int M, int relu) {
        int nwaves = (M + 63) / 64;
        dim3 g((nwaves + 7) / 8, LCOUT[l] / 16);
        pn_gemm_bn_kernel<<<g, 256, 0, stream>>>(X, wh[l], Sv[l], Tv[l], Yh, Yf,
                                                 M, KP[l], LCOUT[l], relu);
    };

    // ---- SA1: N0 -> N1 centers, r=0.2, MLP [32,32,64] ----
    pn_fps_kernel<<<B, FPS_T, 0, stream>>>(points, dist, idx1, N0, N1);
    pn_gather_xyz_kernel<<<(B * N1 + 255) / 256, 256, 0, stream>>>(points, idx1, xyz1, B, N0, N1);
    pn_ball_kernel<<<(B * N1 + 255) / 256, 256, 0, stream>>>(points, xyz1, bal1, B, N0, N1, 0.04f);
    {
        int rows = B * N1 * 16;
        pn_group_kernel<<<(rows + 255) / 256, 256, 0, stream>>>(points, xyz1, bal1, feat0, XA,
                                                                B, N0, N1, 1, KP[0]);
        gemm(XA, 0, XB, nullptr, rows, 1);
        gemm(XB, 1, XA, nullptr, rows, 1);
        gemm(XA, 2, XB, nullptr, rows, 1);
        pn_maxpool_kernel<<<(B * N1 * 64 + 255) / 256, 256, 0, stream>>>(XB, l1p, B * N1, 64);
    }
    // ---- SA2: N1 -> N2 centers, r=0.4, MLP [64,64,128] ----
    pn_fps_kernel<<<B, FPS_T, 0, stream>>>(xyz1, dist, idx2, N1, N2);
    pn_gather_xyz_kernel<<<(B * N2 + 255) / 256, 256, 0, stream>>>(xyz1, idx2, xyz2, B, N1, N2);
    pn_ball_kernel<<<(B * N2 + 255) / 256, 256, 0, stream>>>(xyz1, xyz2, bal2, B, N1, N2, 0.16f);
    {
        int rows = B * N2 * 16;
        pn_group_kernel<<<(rows + 255) / 256, 256, 0, stream>>>(xyz1, xyz2, bal2, l1p, XA,
                                                                B, N1, N2, 64, KP[3]);
        gemm(XA, 3, XB, nullptr, rows, 1);
        gemm(XB, 4, XA, nullptr, rows, 1);
        gemm(XA, 5, XB, nullptr, rows, 1);
        pn_maxpool_kernel<<<(B * N2 * 128 + 255) / 256, 256, 0, stream>>>(XB, l2p, B * N2, 128);
    }
    // ---- SA3: N2 -> N3 centers, r=0.6, MLP [128,128,256] ----
    pn_fps_kernel<<<B, FPS_T, 0, stream>>>(xyz2, dist, idx3, N2, N3);
    pn_gather_xyz_kernel<<<(B * N3 + 255) / 256, 256, 0, stream>>>(xyz2, idx3, xyz3, B, N2, N3);
    pn_ball_kernel<<<(B * N3 + 255) / 256, 256, 0, stream>>>(xyz2, xyz3, bal3, B, N2, N3, 0.36f);
    {
        int rows = B * N3 * 16;
        pn_group_kernel<<<(rows + 255) / 256, 256, 0, stream>>>(xyz2, xyz3, bal3, l2p, XA,
                                                                B, N2, N3, 128, KP[6]);
        gemm(XA, 6, XB, nullptr, rows, 1);
        gemm(XB, 7, XA, nullptr, rows, 1);
        gemm(XA, 8, XB, nullptr, rows, 1);
        pn_maxpool_kernel<<<(B * N3 * 256 + 255) / 256, 256, 0, stream>>>(XB, l3p, B * N3, 256);
    }
    // ---- FP3: target l2 (944), source l3; X = [l2_p(128) | interp(256)] ----
    pn_copy_skip_kernel<<<(B * N2 * 128 + 255) / 256, 256, 0, stream>>>(l2p, XA, B * N2, 128, KP[9], 0);
    pn_interp_kernel<<<(B * N2 + 255) / 256, 256, 0, stream>>>(xyz2, xyz3, l3p, XA,
                                                               B, N2, N3, 256, KP[9], 128);
    gemm(XA, 9, XB, nullptr, B * N2, 1);
    gemm(XB, 10, f3o, nullptr, B * N2, 1);
    // ---- FP2: target l1 (3776), source l2; X = [l1_p(64) | interp(256)] ----
    pn_copy_skip_kernel<<<(B * N1 * 64 + 255) / 256, 256, 0, stream>>>(l1p, XA, B * N1, 64, KP[11], 0);
    pn_interp_kernel<<<(B * N1 + 255) / 256, 256, 0, stream>>>(xyz1, xyz2, f3o, XA,
                                                               B, N1, N2, 256, KP[11], 64);
    gemm(XA, 11, XB, nullptr, B * N1, 1);
    gemm(XB, 12, f2o, nullptr, B * N1, 1);
    // ---- FP1: target l0 (15104), source l1; X = interp(128) ----
    pn_interp_kernel<<<(R0 + 255) / 256, 256, 0, stream>>>(points, xyz1, f2o, XA,
                                                           B, N0, N1, 128, KP[13], 0);
    gemm(XA, 13, XB, nullptr, R0, 1);
    gemm(XB, 14, XA, nullptr, R0, 1);
    gemm(XA, 15, l0p, nullptr, R0, 1);
    // ---- Head ----
    pn_mean_kernel<<<dim3(128, B), 256, 0, stream>>>(l0p, gc, N0, 128);
    pn_headx_kernel<<<((unsigned)((size_t)R0 * 128 + 255) / 256), 256, 0, stream>>>(l0p, gc, XB, B, N0, 128);
    gemm(XB, 16, nullptr, h1o, R0, 0);  // head1, relu=False, f32 out
    pn_head2_kernel<<<(R0 + 255) / 256, 256, 0, stream>>>(h1o, (const float*)d_in[104],
                                                          (const float*)d_in[105],
                                                          points, (float*)d_out, R0, 128);
}
